// RuleAugmentedCOFRN_32830730010729
// MI455X (gfx1250) — matrix operand order repaired and checked by reading, not compiled, not run on hardware
//
#include <hip/hip_runtime.h>
#include <math.h>

typedef __attribute__((ext_vector_type(16))) __bf16 v16bf;
typedef __attribute__((ext_vector_type(8)))  __bf16 v8bf;
typedef __attribute__((ext_vector_type(8)))  float  v8f;

#define DIM 384           // D
#define HID 256           // H
#define KSTEPS 12         // 384 / 32
#define TOPK 8
#define NSLICES 8         // N-dimension split for occupancy
#define NEG_INF (-__builtin_inff())

// async global -> LDS copy (ASYNCcnt); la = LDS byte offset, ga = global addr
#define ASYNC_LD_B128(la, ga)                                                  \
  asm volatile("global_load_async_to_lds_b128 %0, %1, off"                     \
               :: "v"(la), "v"(ga) : "memory")
#define WAIT_ASYNC0() asm volatile("s_wait_asynccnt 0x0" ::: "memory")

__device__ __forceinline__ v16bf combine16(v8bf lo, v8bf hi) {
  v16bf r;
#pragma unroll
  for (int i = 0; i < 8; ++i) { r[i] = lo[i]; r[i + 8] = hi[i]; }
  return r;
}

// ---- DPP butterfly helpers (confined to rows of 16 lanes; pure VALU) ----
#define DPP_F(x, CTRL)                                                         \
  __builtin_bit_cast(float, __builtin_amdgcn_update_dpp(                       \
      __builtin_bit_cast(int, x), __builtin_bit_cast(int, x), CTRL, 0xF, 0xF, false))

__device__ __forceinline__ float row16_max(float x) {
  x = fmaxf(x, DPP_F(x, 0xB1));
  x = fmaxf(x, DPP_F(x, 0x4E));
  x = fmaxf(x, DPP_F(x, 0x141));
  x = fmaxf(x, DPP_F(x, 0x140));
  return x;
}

__device__ __forceinline__ float row16_sum(float x) {
  x += DPP_F(x, 0xB1);
  x += DPP_F(x, 0x4E);
  x += DPP_F(x, 0x141);
  x += DPP_F(x, 0x140);
  return x;
}

#define ARGMAX_STEP(CTRL)                                                      \
  {                                                                            \
    float _ov = DPP_F(mv, CTRL);                                               \
    int _os = __builtin_amdgcn_update_dpp(src, src, CTRL, 0xF, 0xF, false);    \
    if (_ov > mv || (_ov == mv && _os < src)) { mv = _ov; src = _os; }         \
  }

// ---------------- row L2-normalize (f32 -> bf16), one wave per row --------
__global__ __launch_bounds__(256) void rn_normalize_rows(const float* __restrict__ in,
                                                         __bf16* __restrict__ outb,
                                                         int rows) {
  const int lane = threadIdx.x & 31;
  const int row  = blockIdx.x * 8 + (threadIdx.x >> 5);
  if (row >= rows) return;
  const float* p = in + (size_t)row * DIM;
  float ss = 0.f;
#pragma unroll
  for (int d = lane; d < DIM; d += 32) { float v = p[d]; ss += v * v; }
#pragma unroll
  for (int off = 16; off >= 1; off >>= 1) ss += __shfl_xor(ss, off, 32);
  const float scale = 1.0f / fmaxf(sqrtf(ss), 1e-12f);
  __bf16* o = outb + (size_t)row * DIM;
#pragma unroll
  for (int d = lane; d < DIM; d += 32) o[d] = (__bf16)(p[d] * scale);
}

// ---------------- f32 -> bf16 elementwise -------------------------------
__global__ __launch_bounds__(256) void rn_f2bf(const float* __restrict__ in,
                                               __bf16* __restrict__ out, int n) {
  int i = blockIdx.x * 256 + threadIdx.x;
  if (i < n) out[i] = (__bf16)in[i];
}

// ---------------- fused sims-GEMM + streaming partial top-8 ---------------
// Grid: (B/128, NSLICES). WG = 8 waves, each wave owns 16 rows of q.
// B tiles stream into double-buffered LDS via async-load (ASYNCcnt), so the
// WMMA pipeline's DScnt waits only ever cover fragment ds_loads.
__global__ __launch_bounds__(256) void rn_topk_partial(const __bf16* __restrict__ qbf,
                                                       const __bf16* __restrict__ rbf,
                                                       float* __restrict__ pv,
                                                       int* __restrict__ pi,
                                                       int B, int nChunks, int cps) {
  const int lane = threadIdx.x & 31;
  const int wid  = threadIdx.x >> 5;
  const int mBase = (blockIdx.x * 8 + wid) * 16;
  const int slice = blockIdx.y;
  const int cBeg = slice * cps;
  const int cEnd = (cBeg + cps < nChunks) ? (cBeg + cps) : nChunks;

  __shared__ __align__(16) __bf16 sB[2][16 * 392];   // double-buffered B tile
  __shared__ float sTopV[8][16][TOPK];
  __shared__ int   sTopI[8][16][TOPK];
  __shared__ float sMin[8][16];

  for (int i = lane; i < 16 * TOPK; i += 32) {
    sTopV[wid][i >> 3][i & 7] = NEG_INF;
    sTopI[wid][i >> 3][i & 7] = 0;
  }
  if (lane < 16) sMin[wid][lane] = NEG_INF;

  // preload A fragments (wave's 16 q rows, K = 384)
  v16bf afrag[KSTEPS];
  {
    const __bf16* abase = qbf + (size_t)(mBase + (lane & 15)) * DIM + ((lane >> 4) << 3);
#pragma unroll
    for (int ks = 0; ks < KSTEPS; ++ks) {
      const __bf16* p = abase + ks * 32;
      afrag[ks] = combine16(*(const v8bf*)p, *(const v8bf*)(p + 16));
    }
  }

  // staging coords: 768 x 16B chunks per tile, 3 per thread
  int scol[3], soff[3];
  unsigned int lds0[3], lds1[3];
#pragma unroll
  for (int j = 0; j < 3; ++j) {
    int c = threadIdx.x + j * 256;
    scol[j] = c / 48;
    soff[j] = (c % 48) * 8;
    int lo = scol[j] * 392 + soff[j];
    lds0[j] = (unsigned int)(size_t)&sB[0][lo];
    lds1[j] = (unsigned int)(size_t)&sB[1][lo];
  }

  // prologue: async-load chunk cBeg into its buffer
#pragma unroll
  for (int j = 0; j < 3; ++j) {
    unsigned long long ga =
        (unsigned long long)(size_t)&rbf[(size_t)(cBeg * 16 + scol[j]) * DIM + soff[j]];
    ASYNC_LD_B128((cBeg & 1) ? lds1[j] : lds0[j], ga);
  }

  float rmin[8];
#pragma unroll
  for (int r = 0; r < 8; ++r) rmin[r] = NEG_INF;

  const int bcol  = lane & 15;
  const int bkh   = (lane >> 4) << 3;
  const int rhalf = (lane >> 4) << 3;

  for (int it = cBeg; it < cEnd; ++it) {
    WAIT_ASYNC0();        // my async writes to LDS landed
    __syncthreads();      // everyone's landed

    if (it + 1 < cEnd) {  // kick off next chunk into the other buffer
      const int nNext16 = (it + 1) * 16;
#pragma unroll
      for (int j = 0; j < 3; ++j) {
        unsigned long long ga =
            (unsigned long long)(size_t)&rbf[(size_t)(nNext16 + scol[j]) * DIM + soff[j]];
        ASYNC_LD_B128(((it + 1) & 1) ? lds1[j] : lds0[j], ga);
      }
    }

    // preload ALL 12 B fragments, fence the scheduler, then 12 WMMAs
    const __bf16* bbase = &sB[it & 1][bcol * 392 + bkh];
    v16bf bfr[KSTEPS];
#pragma unroll
    for (int ks = 0; ks < KSTEPS; ++ks) {
      const __bf16* p = bbase + ks * 32;
      bfr[ks] = combine16(*(const v8bf*)p, *(const v8bf*)(p + 16));
    }
    __builtin_amdgcn_sched_barrier(0);
    v8f acc = {};
#pragma unroll
    for (int ks = 0; ks < KSTEPS; ++ks)
      acc = __builtin_amdgcn_wmma_f32_16x16x32_bf16(false, afrag[ks], false, bfr[ks],
                                                    (short)0, acc, false, false);

    // combined fast check: any candidate beating its row's 8th-best?
    float cand = acc[0] - rmin[0];
#pragma unroll
    for (int r = 1; r < 8; ++r) cand = fmaxf(cand, acc[r] - rmin[r]);
    if (row16_max(cand) > 0.0f) {
      const int nBase = it * 16;
#pragma unroll
      for (int r = 0; r < 8; ++r) {
        float v = acc[r];
        const int rowL = r + rhalf;
        while (true) {
          float mv = v; int src = lane;
          ARGMAX_STEP(0xB1)
          ARGMAX_STEP(0x4E)
          ARGMAX_STEP(0x141)
          ARGMAX_STEP(0x140)
          if (mv <= rmin[r]) break;
          if (lane == src) {
            float* tv = sTopV[wid][rowL];
            int*   ti = sTopI[wid][rowL];
            int j = TOPK - 1;
            while (j > 0 && tv[j - 1] < v) { tv[j] = tv[j - 1]; ti[j] = ti[j - 1]; --j; }
            tv[j] = v; ti[j] = nBase + bcol;
            sMin[wid][rowL] = tv[TOPK - 1];
            v = NEG_INF;
          }
          rmin[r] = sMin[wid][rowL];
        }
      }
    }
  }

  // emit partial top-8: pv/pi layout [NSLICES][B][8]
  for (int i = lane; i < 16 * TOPK; i += 32) {
    const int row = i >> 3, k = i & 7;
    const size_t o = ((size_t)slice * B + mBase + row) * TOPK + k;
    pv[o] = sTopV[wid][row][k];
    pi[o] = sTopI[wid][row][k];
  }
}

// ---------------- merge partial top-8s, softmax, gather context ----------
__global__ __launch_bounds__(256) void rn_topk_merge(const float* __restrict__ pv,
                                                     const int* __restrict__ pi,
                                                     const __bf16* __restrict__ rbf,
                                                     __bf16* __restrict__ ctxbf,
                                                     int B) {
  const int lane = threadIdx.x & 31;
  const int wid  = threadIdx.x >> 5;
  const int row  = blockIdx.x * 8 + wid;

  __shared__ float cv[8][NSLICES * TOPK];
  __shared__ int   ci[8][NSLICES * TOPK];
  __shared__ float outv[8][TOPK];
  __shared__ int   outi[8][TOPK];

  for (int i = lane; i < NSLICES * TOPK; i += 32) {
    const int s = i >> 3, k = i & 7;
    const size_t o = ((size_t)s * B + row) * TOPK + k;
    cv[wid][i] = pv[o];
    ci[wid][i] = pi[o];
  }
  if (lane == 0) {
    float tv[TOPK]; int ti[TOPK];
#pragma unroll
    for (int k = 0; k < TOPK; ++k) { tv[k] = NEG_INF; ti[k] = 0; }
    for (int i = 0; i < NSLICES * TOPK; ++i) {
      float v = cv[wid][i];
      if (v > tv[TOPK - 1]) {
        int idx = ci[wid][i];
        int j = TOPK - 1;
        while (j > 0 && tv[j - 1] < v) { tv[j] = tv[j - 1]; ti[j] = ti[j - 1]; --j; }
        tv[j] = v; ti[j] = idx;
      }
    }
#pragma unroll
    for (int k = 0; k < TOPK; ++k) { outv[wid][k] = tv[k]; outi[wid][k] = ti[k]; }
  }

  float w[TOPK]; int idx[TOPK];
  const float mx = outv[wid][0];
  float sum = 0.f;
#pragma unroll
  for (int k = 0; k < TOPK; ++k) {
    w[k] = __expf(outv[wid][k] - mx);
    sum += w[k];
    idx[k] = outi[wid][k];
  }
  const float inv = 1.0f / sum;
  for (int d = lane; d < DIM; d += 32) {
    float a = 0.f;
#pragma unroll
    for (int k = 0; k < TOPK; ++k) a += (w[k] * inv) * (float)rbf[(size_t)idx[k] * DIM + d];
    ctxbf[(size_t)row * DIM + d] = (__bf16)a;
  }
}

// ---------------- WMMA injection GEMM + GELU + residual + LayerNorm ------
__global__ __launch_bounds__(256) void rn_inject_ln(const __bf16* __restrict__ ctxbf,
                                                    const float* __restrict__ s0,
                                                    const __bf16* __restrict__ wbf,
                                                    const float* __restrict__ bvec,
                                                    const float* __restrict__ alphap,
                                                    const float* __restrict__ gamma,
                                                    const float* __restrict__ beta,
                                                    float* __restrict__ out) {
  const int lane = threadIdx.x & 31;
  const int wid  = threadIdx.x >> 5;
  const int mBase = blockIdx.x * 16;

  __shared__ float sx[16][HID];

  v16bf afrag[KSTEPS];
  {
    const __bf16* abase = ctxbf + (size_t)(mBase + (lane & 15)) * DIM + ((lane >> 4) << 3);
#pragma unroll
    for (int ks = 0; ks < KSTEPS; ++ks) {
      const __bf16* p = abase + ks * 32;
      afrag[ks] = combine16(*(const v8bf*)p, *(const v8bf*)(p + 16));
    }
  }

  const float alpha = *alphap;
  const int bkh = (lane >> 4) << 3;
  const int rhalf = (lane >> 4) << 3;

#pragma unroll
  for (int t = 0; t < 2; ++t) {
    const int hBase = (wid * 2 + t) * 16;
    const int col = hBase + (lane & 15);
    const __bf16* bbase = wbf + (size_t)col * DIM + bkh;
    v16bf bfr[KSTEPS];
#pragma unroll
    for (int ks = 0; ks < KSTEPS; ++ks) {
      const __bf16* p = bbase + ks * 32;
      bfr[ks] = combine16(*(const v8bf*)p, *(const v8bf*)(p + 16));
    }
    __builtin_amdgcn_sched_barrier(0);
    v8f acc = {};
#pragma unroll
    for (int ks = 0; ks < KSTEPS; ++ks)
      acc = __builtin_amdgcn_wmma_f32_16x16x32_bf16(false, afrag[ks], false, bfr[ks],
                                                    (short)0, acc, false, false);
#pragma unroll
    for (int r = 0; r < 8; ++r) {
      const int rowL = r + rhalf;
      float pre = acc[r] + bvec[col];
      float g = 0.5f * pre * (1.0f + erff(pre * 0.70710678118654752f));
      sx[rowL][col] = s0[(size_t)(mBase + rowL) * HID + col] + alpha * g;
    }
  }
  __syncthreads();

  const int rowL = threadIdx.x >> 4;
  const int sub  = threadIdx.x & 15;
  float sum = 0.f, sq = 0.f;
#pragma unroll
  for (int j = 0; j < 16; ++j) {
    float xv = sx[rowL][sub + 16 * j];
    sum += xv; sq += xv * xv;
  }
  sum = row16_sum(sum);
  sq  = row16_sum(sq);
  const float mu = sum * (1.0f / HID);
  const float var = sq * (1.0f / HID) - mu * mu;
  const float rstd = rsqrtf(var + 1e-5f);
#pragma unroll
  for (int j = 0; j < 16; ++j) {
    const int col = sub + 16 * j;
    float xv = sx[rowL][col];
    out[(size_t)(mBase + rowL) * HID + col] = (xv - mu) * rstd * gamma[col] + beta[col];
  }
}

// -------------------------------------------------------------------------
extern "C" void kernel_launch(void* const* d_in, const int* in_sizes, int n_in,
                              void* d_out, int out_size, void* d_ws, size_t ws_size,
                              hipStream_t stream) {
  const float* emb    = (const float*)d_in[0];   // [B, 384]
  const float* s0     = (const float*)d_in[1];   // [B, 256]
  const float* rules  = (const float*)d_in[2];   // [N, 384]
  const float* W      = (const float*)d_in[3];   // [256, 384]
  const float* bvec   = (const float*)d_in[4];   // [256]
  const float* alphap = (const float*)d_in[5];   // scalar
  const float* gamma  = (const float*)d_in[6];   // [256]
  const float* beta   = (const float*)d_in[7];   // [256]
  // d_in[8] = top_k (fixed at 8)

  const int B = in_sizes[0] / DIM;
  const int N = in_sizes[2] / DIM;
  const int nChunks = N / 16;
  const int cps = (nChunks + NSLICES - 1) / NSLICES;

  char* ws = (char*)d_ws;
  size_t off = 0;
  __bf16* rbf = (__bf16*)(ws + off); off += ((size_t)N * DIM * 2 + 255) & ~(size_t)255;
  __bf16* qbf = (__bf16*)(ws + off); off += ((size_t)B * DIM * 2 + 255) & ~(size_t)255;
  __bf16* ctxbf = (__bf16*)(ws + off); off += ((size_t)B * DIM * 2 + 255) & ~(size_t)255;
  __bf16* wbf = (__bf16*)(ws + off); off += ((size_t)HID * DIM * 2 + 255) & ~(size_t)255;
  float* pv = (float*)(ws + off); off += ((size_t)NSLICES * B * TOPK * 4 + 255) & ~(size_t)255;
  int* pi = (int*)(ws + off); off += ((size_t)NSLICES * B * TOPK * 4 + 255) & ~(size_t)255;
  (void)ws_size; (void)n_in; (void)out_size;

  rn_normalize_rows<<<(N + 7) / 8, 256, 0, stream>>>(rules, rbf, N);
  rn_normalize_rows<<<(B + 7) / 8, 256, 0, stream>>>(emb, qbf, B);
  {
    int n = HID * DIM;
    rn_f2bf<<<(n + 255) / 256, 256, 0, stream>>>(W, wbf, n);
  }
  rn_topk_partial<<<dim3(B / 128, NSLICES), 256, 0, stream>>>(qbf, rbf, pv, pi,
                                                              B, nChunks, cps);
  rn_topk_merge<<<B / 8, 256, 0, stream>>>(pv, pi, rbf, ctxbf, B);
  rn_inject_ln<<<B / 16, 256, 0, stream>>>(ctxbf, s0, wbf, bvec, alphap, gamma, beta,
                                           (float*)d_out);
}
